// TrainableReservoir_2302102470817
// MI455X (gfx1250) — matrix operand order repaired and checked by reading, not compiled, not run on hardware
//
#include <hip/hip_runtime.h>
#include <math.h>

#define ALPHA 0.5f
#define BK 128                     // k-chunk depth (4 WMMA k-steps)

typedef __attribute__((ext_vector_type(16))) __bf16 v16bf;
typedef __attribute__((ext_vector_type(8)))  float  v8f;

__device__ __forceinline__ unsigned short f32_to_bf16_rne(float f) {
    unsigned int u = __float_as_uint(f);
    u += 0x7FFFu + ((u >> 16) & 1u);
    return (unsigned short)(u >> 16);
}

// tanh(x) = 1 - 2/(exp2(2*log2e*x)+1)  -> v_exp_f32 + v_rcp_f32, exact at +/-inf
__device__ __forceinline__ float fast_tanh(float x) {
    const float e = __builtin_amdgcn_exp2f(x * 2.8853900817779268f);
    return 1.0f - 2.0f * __builtin_amdgcn_rcpf(e + 1.0f);
}

union Frag16 {
    uint4 q[2];
    unsigned short s[16];
    v16bf v;
};

// ---------------------------------------------------------------------------
// fp32 -> bf16 conversion, 8 elements / thread (n must be a multiple of 8)
// ---------------------------------------------------------------------------
__global__ void __launch_bounds__(256)
cvt_f32_bf16_x8(const float* __restrict__ src, unsigned short* __restrict__ dst, int n8) {
    int i = blockIdx.x * blockDim.x + threadIdx.x;
    if (i >= n8) return;
    const float4 f0 = ((const float4*)src)[2 * i];
    const float4 f1 = ((const float4*)src)[2 * i + 1];
    union { unsigned short s[8]; uint4 q; } o;
    o.s[0] = f32_to_bf16_rne(f0.x); o.s[1] = f32_to_bf16_rne(f0.y);
    o.s[2] = f32_to_bf16_rne(f0.z); o.s[3] = f32_to_bf16_rne(f0.w);
    o.s[4] = f32_to_bf16_rne(f1.x); o.s[5] = f32_to_bf16_rne(f1.y);
    o.s[6] = f32_to_bf16_rne(f1.z); o.s[7] = f32_to_bf16_rne(f1.w);
    ((uint4*)dst)[i] = o.q;
}

__global__ void __launch_bounds__(256)
zero_u16(unsigned short* __restrict__ dst, int n) {
    int i = blockIdx.x * blockDim.x + threadIdx.x;
    if (i < n) dst[i] = (unsigned short)0;
}

// ---------------------------------------------------------------------------
// Issue async global->LDS copies for one 128-deep k-chunk (no wait here):
//   A panel: 64  rows x 128 cols bf16 (16 KB) -> 1024 b128 ops, 4 / thread
//   B panel: 128 rows x 128 cols bf16 (32 KB) -> 2048 b128 ops, 8 / thread
// ---------------------------------------------------------------------------
__device__ __forceinline__ void issue_panels(
    unsigned short* sA, unsigned short* sB,
    const unsigned short* __restrict__ Ag, const unsigned short* __restrict__ Bg,
    int K, int m0blk, int n0blk, int k0, int tid)
{
    #pragma unroll
    for (int s = 0; s < 4; ++s) {
        const int idx = tid + s * 256;
        const int r = idx >> 4, part = idx & 15;
        const unsigned short* src = Ag + (size_t)(m0blk + r) * K + k0 + part * 8;
        const unsigned dst = (unsigned)(size_t)sA + (unsigned)(r * 256 + part * 16);
        asm volatile("global_load_async_to_lds_b128 %0, %1, off"
                     :: "v"(dst), "v"(src) : "memory");
    }
    #pragma unroll
    for (int s = 0; s < 8; ++s) {
        const int idx = tid + s * 256;
        const int r = idx >> 4, part = idx & 15;
        const unsigned short* src = Bg + (size_t)(n0blk + r) * K + k0 + part * 8;
        const unsigned dst = (unsigned)(size_t)sB + (unsigned)(r * 256 + part * 16);
        asm volatile("global_load_async_to_lds_b128 %0, %1, off"
                     :: "v"(dst), "v"(src) : "memory");
    }
}

// ---------------------------------------------------------------------------
// Double-buffered block-tile mainloop: 64x128 output per block, 8 waves.
//   wave w: m-tiles = 2*(w&1) + {0,1}, n-tiles = 2*(w>>1) + {0,1}
// acc[mi][j] accumulates the corresponding 16x16 tile (2x2 quad per wave).
// Pipeline per iter: wait(my buf-i loads) ; barrier ; issue buf-(i+1) ; compute.
// The single barrier both publishes buf-i and proves buf-(i+1)'s slot is free.
// ---------------------------------------------------------------------------
__device__ __forceinline__ void gemm_mainloop(
    unsigned short (*sA)[64 * BK], unsigned short (*sB)[128 * BK],
    const unsigned short* __restrict__ Ag, const unsigned short* __restrict__ Bg,
    int K, int m0blk, int n0blk, v8f acc[2][2])
{
    const int tid  = threadIdx.x;
    const int lane = tid & 31;
    const int wave = tid >> 5;
    const int half = lane >> 4;
    const int l15  = lane & 15;
    const int mgrp = wave & 1;        // m-tiles 2*mgrp, 2*mgrp+1
    const int ngrp = wave >> 1;       // n-tiles 2*ngrp, 2*ngrp+1

    const int nk = K >> 7;
    issue_panels(sA[0], sB[0], Ag, Bg, K, m0blk, n0blk, 0, tid);

    for (int i = 0; i < nk; ++i) {
        asm volatile("s_wait_asynccnt 0x0" ::: "memory");
        __syncthreads();
        if (i + 1 < nk)
            issue_panels(sA[(i + 1) & 1], sB[(i + 1) & 1],
                         Ag, Bg, K, m0blk, n0blk, (i + 1) << 7, tid);

        const unsigned short* cA = sA[i & 1];
        const unsigned short* cB = sB[i & 1];

        #pragma unroll
        for (int kk = 0; kk < 4; ++kk) {
            Frag16 a[2];
            #pragma unroll
            for (int mi = 0; mi < 2; ++mi) {
                const unsigned short* ar =
                    cA + ((mgrp * 2 + mi) * 16 + l15) * BK + kk * 32;
                a[mi].q[0] = *(const uint4*)(ar + half * 8);
                a[mi].q[1] = *(const uint4*)(ar + 16 + half * 8);
            }
            #pragma unroll
            for (int j = 0; j < 2; ++j) {
                Frag16 b;
                const unsigned short* br =
                    cB + ((ngrp * 2 + j) * 16 + l15) * BK + kk * 32;
                b.q[0] = *(const uint4*)(br + half * 8);
                b.q[1] = *(const uint4*)(br + 16 + half * 8);
                #pragma unroll
                for (int mi = 0; mi < 2; ++mi) {
                    acc[mi][j] = __builtin_amdgcn_wmma_f32_16x16x32_bf16(
                        false, a[mi].v, false, b.v, (short)0, acc[mi][j],
                        false, false);
                }
            }
        }
    }
}

// ---------------------------------------------------------------------------
// Phase 1: out[row, n] = sum_d Xb[row,d] * Winb[n,d] + bias[n]
// ---------------------------------------------------------------------------
__global__ void __launch_bounds__(256)
xproj_gemm(const unsigned short* __restrict__ Xb, const unsigned short* __restrict__ Winb,
           const float* __restrict__ bias, float* __restrict__ out,
           int M, int N, int K) {
    __shared__ unsigned short sA[2][64 * BK];
    __shared__ unsigned short sB[2][128 * BK];

    const int nBlk  = N >> 7;
    const int m0blk = (blockIdx.x / nBlk) << 6;
    const int n0blk = (blockIdx.x % nBlk) << 7;

    v8f acc[2][2];
    #pragma unroll
    for (int mi = 0; mi < 2; ++mi)
        #pragma unroll
        for (int j = 0; j < 2; ++j)
            #pragma unroll
            for (int i = 0; i < 8; ++i) acc[mi][j][i] = 0.0f;

    gemm_mainloop(sA, sB, Xb, Winb, K, m0blk, n0blk, acc);

    const int lane = threadIdx.x & 31;
    const int wave = threadIdx.x >> 5;
    const int half = lane >> 4;
    const int l15  = lane & 15;
    const int mgrp = wave & 1;
    const int ngrp = wave >> 1;

    #pragma unroll
    for (int mi = 0; mi < 2; ++mi) {
        #pragma unroll
        for (int j = 0; j < 2; ++j) {
            const int n = n0blk + (ngrp * 2 + j) * 16 + l15;
            const float bv = bias[n];
            #pragma unroll
            for (int i = 0; i < 8; ++i) {
                const int m = m0blk + (mgrp * 2 + mi) * 16 + i + half * 8;
                out[(size_t)m * N + n] = acc[mi][j][i] + bv;
            }
        }
    }
}

// ---------------------------------------------------------------------------
// Phase 2: one recurrence step (M = 64 batch rows, K = N = 2048)
// ---------------------------------------------------------------------------
__global__ void __launch_bounds__(256)
reservoir_step(const unsigned short* __restrict__ Rprev_bf,
               const unsigned short* __restrict__ Wb,
               float* __restrict__ out,
               unsigned short* __restrict__ Rnext_bf,
               int t, int T, int N) {
    __shared__ unsigned short sA[2][64 * BK];
    __shared__ unsigned short sB[2][128 * BK];

    const int n0blk = blockIdx.x << 7;
    const int m0blk = 0;

    v8f acc[2][2];
    #pragma unroll
    for (int mi = 0; mi < 2; ++mi)
        #pragma unroll
        for (int j = 0; j < 2; ++j)
            #pragma unroll
            for (int i = 0; i < 8; ++i) acc[mi][j][i] = 0.0f;

    gemm_mainloop(sA, sB, Rprev_bf, Wb, N, m0blk, n0blk, acc);

    const int lane = threadIdx.x & 31;
    const int wave = threadIdx.x >> 5;
    const int half = lane >> 4;
    const int l15  = lane & 15;
    const int mgrp = wave & 1;
    const int ngrp = wave >> 1;

    #pragma unroll
    for (int mi = 0; mi < 2; ++mi) {
        #pragma unroll
        for (int j = 0; j < 2; ++j) {
            const int n = n0blk + (ngrp * 2 + j) * 16 + l15;
            #pragma unroll
            for (int i = 0; i < 8; ++i) {
                const int m = (mgrp * 2 + mi) * 16 + i + half * 8;   // batch index
                const size_t orow = ((size_t)m * T + t) * (size_t)N + n;
                const float xp = out[orow];
                const float rp = (t > 0) ? out[orow - N] : 0.0f;
                const float pre = acc[mi][j][i] + xp;
                const float r = (1.0f - ALPHA) * rp + ALPHA * fast_tanh(pre);
                out[orow] = r;
                Rnext_bf[(size_t)m * N + n] = f32_to_bf16_rne(r);
            }
        }
    }
}

// ---------------------------------------------------------------------------
extern "C" void kernel_launch(void* const* d_in, const int* in_sizes, int n_in,
                              void* d_out, int out_size, void* d_ws, size_t ws_size,
                              hipStream_t stream) {
    const float* x_seq = (const float*)d_in[0];   // (B,T,D)
    const float* W_in  = (const float*)d_in[1];   // (N,D)
    const float* W     = (const float*)d_in[2];   // (N,N)
    const float* bias  = (const float*)d_in[3];   // (N,)
    float* out = (float*)d_out;                   // (B,T,N)

    const int B = 64, T = 512, D = 256, N = 2048;

    unsigned char* ws = (unsigned char*)d_ws;
    size_t off = 0;
    unsigned short* Wb   = (unsigned short*)(ws + off); off += (size_t)N * N * 2;      // 8 MB
    unsigned short* Winb = (unsigned short*)(ws + off); off += (size_t)N * D * 2;      // 1 MB
    unsigned short* Xb   = (unsigned short*)(ws + off); off += (size_t)B * T * D * 2;  // 16 MB
    unsigned short* rA   = (unsigned short*)(ws + off); off += (size_t)B * N * 2;
    unsigned short* rB   = (unsigned short*)(ws + off);

    // per-call conversions + state init
    cvt_f32_bf16_x8<<<(N * N / 8 + 255) / 256, 256, 0, stream>>>(W, Wb, N * N / 8);
    cvt_f32_bf16_x8<<<(N * D / 8 + 255) / 256, 256, 0, stream>>>(W_in, Winb, N * D / 8);
    cvt_f32_bf16_x8<<<(B * T * D / 8 + 255) / 256, 256, 0, stream>>>(x_seq, Xb, B * T * D / 8);
    zero_u16<<<(B * N + 255) / 256, 256, 0, stream>>>(rA, B * N);

    // phase 1: x_proj (+bias) into d_out
    const int M1 = B * T;                              // 32768
    const int blocks1 = (M1 / 64) * (N / 128);         // 8192
    xproj_gemm<<<blocks1, 256, 0, stream>>>(Xb, Winb, bias, out, M1, N, D);

    // phase 2: sequential recurrence
    const int blocks2 = (B / 64) * (N / 128);          // 16
    for (int t = 0; t < T; ++t) {
        unsigned short* rp = (t & 1) ? rB : rA;
        unsigned short* rn = (t & 1) ? rA : rB;
        reservoir_step<<<blocks2, 256, 0, stream>>>(rp, Wb, out, rn, t, T, N);
    }
}
